// DIGQ_91319594648354
// MI455X (gfx1250) — matrix operand order repaired and checked by reading, not compiled, not run on hardware
//
#include <hip/hip_runtime.h>
#include <hip/hip_bf16.h>

typedef __attribute__((ext_vector_type(16))) _Float16 v16h;
typedef __attribute__((ext_vector_type(8)))  _Float16 v8h;
typedef __attribute__((ext_vector_type(8)))  float    v8f;
typedef int v4i __attribute__((vector_size(16)));
typedef __attribute__((address_space(1))) v4i g_v4i;
typedef __attribute__((address_space(3))) v4i l_v4i;

// ---------------------------------------------------------------------------
// Problem constants
// ---------------------------------------------------------------------------
#define BSZ   16
#define SEQ   256
#define NAG   8
#define OBS   64
#define SD    128
#define EMB   128
#define HD    4
#define NL    2
#define BTOT  (BSZ*SEQ)          // 4096
#define ROWS  (BTOT*NAG)         // 32768 node rows
#define KNF   96                 // 65 padded to multiple of 32
#define KCAT  (EMB*(1+HD))       // 640
#define KQP   (NAG*EMB)          // 1024
#define WPAD  8                  // LDS row padding (halves) -> bank-clean

// Async LDS-copy path (probe-verified on this toolchain)
#if defined(__has_builtin)
#  if __has_builtin(__builtin_amdgcn_global_load_async_to_lds_b128) && \
      __has_builtin(__builtin_amdgcn_s_wait_asynccnt)
#    define HAVE_ASYNC_LDS 1
#  endif
#endif
#ifndef HAVE_ASYNC_LDS
#  define HAVE_ASYNC_LDS 0
#endif

__device__ __forceinline__ void async_wait_all()
{
#if HAVE_ASYNC_LDS
    __builtin_amdgcn_s_wait_asynccnt(0);
#endif
}

// fragment loaders ----------------------------------------------------------
__device__ __forceinline__ v16h load_frag_a(const _Float16* p)   // gaps 0..7,16..23
{
    v8h lo = *(const v8h*)p;
    v8h hi = *(const v8h*)(p + 16);
    return __builtin_shufflevector(lo, hi, 0,1,2,3,4,5,6,7,8,9,10,11,12,13,14,15);
}
__device__ __forceinline__ v16h load_frag_b(const _Float16* p)   // gaps 0..7,16..23
{
    v8h lo = *(const v8h*)p;
    v8h hi = *(const v8h*)(p + 16);
    return __builtin_shufflevector(lo, hi, 0,1,2,3,4,5,6,7,8,9,10,11,12,13,14,15);
}

// 8 N-tiles of WMMA against one A fragment; 2-deep B pipeline
__device__ __forceinline__ void mma_row8(v8f* acc, v16h a,
                                         const _Float16* bbase, int WROW)
{
    v16h bc = load_frag_b(bbase);
    #pragma unroll
    for (int nt = 0; nt < 8; ++nt) {
        v16h bn = bc;
        if (nt < 7) bn = load_frag_b(bbase + (nt + 1) * 16 * WROW);
        acc[nt] = __builtin_amdgcn_wmma_f32_16x16x32_f16(
                      false, a, false, bc, (short)0, acc[nt], false, false);
        bc = bn;
    }
}

// stage one K-chunk of pre-transposed f16 weights into (bank-padded) LDS
__device__ __forceinline__ void stage_chunk(const _Float16* __restrict__ Wg,
                                            _Float16* sbuf, int kc, int Kpad,
                                            int KC8, int WROW, int tid)
{
    for (int idx = tid; idx < 128 * KC8; idx += 256) {
        int n  = idx / KC8;
        int ko = (idx - n * KC8) * 8;
        const _Float16* src = Wg + (size_t)n * Kpad + kc + ko;
        _Float16*       dst = sbuf + n * WROW + ko;
#if HAVE_ASYNC_LDS
        __builtin_amdgcn_global_load_async_to_lds_b128(
            (g_v4i*)src, (l_v4i*)dst, 0, 0);
#else
        *(v8h*)dst = *(const v8h*)src;
#endif
    }
}

// ---------------------------------------------------------------------------
// prep kernels
// ---------------------------------------------------------------------------
__global__ __launch_bounds__(256) void prep_nf(const float* __restrict__ qs,
                                               const float* __restrict__ obs,
                                               _Float16* __restrict__ nf)
{
    int idx = blockIdx.x * 256 + threadIdx.x;
    if (idx >= ROWS * KNF) return;
    int row = idx / KNF;
    int k   = idx - row * KNF;
    float v = 0.f;
    if (k == 0)       v = qs[row];
    else if (k < 65)  v = obs[(size_t)row * OBS + (k - 1)];
    nf[idx] = (_Float16)v;
}

__global__ __launch_bounds__(256) void prep_st(const float* __restrict__ st,
                                               _Float16* __restrict__ sth)
{
    int idx = blockIdx.x * 256 + threadIdx.x;
    if (idx >= BTOT * SD) return;
    sth[idx] = (_Float16)st[idx];
}

// one-time: W[Kreal x 128] f32 -> Wt[128 x Kpad] f16 (transposed, padded)
__global__ __launch_bounds__(256) void wprep(const float* __restrict__ W,
                                             _Float16* __restrict__ Wt,
                                             int Kreal, int Kpad)
{
    int idx = blockIdx.x * 256 + threadIdx.x;
    if (idx >= 128 * Kpad) return;
    int k = idx >> 7;
    int n = idx & 127;
    float v = (k < Kreal) ? W[(size_t)k * 128 + n] : 0.f;
    Wt[(size_t)n * Kpad + k] = (_Float16)v;
}

// ---------------------------------------------------------------------------
// WMMA GEMM:  C[M,128] = A[M,Kpad](f16) * Wt + bias (opt relu/res/outF/outH)
// ---------------------------------------------------------------------------
__global__ __launch_bounds__(256) void gemm128(
    const _Float16* __restrict__ A,
    const _Float16* __restrict__ Wt,
    const float*    __restrict__ bias,
    const float*    __restrict__ res,
    float*          __restrict__ outF,
    _Float16*       __restrict__ outH,
    int Kpad, int relu)
{
    extern __shared__ _Float16 smem[];

    const int tid   = threadIdx.x;
    const int lane  = tid & 31;
    const int wave  = tid >> 5;
    const int khi   = lane >> 4;
    const int lan16 = lane & 15;
    const int row0  = blockIdx.x * 128 + wave * 16;

    const int Kc      = (Kpad < 128) ? Kpad : 128;
    const int KC8     = Kc >> 3;
    const int WROW    = Kc + WPAD;
    const int nchunks = Kpad / Kc;
    _Float16* buf0 = smem;
    _Float16* buf1 = smem + 128 * WROW;

    v8f acc[8] = {};

    stage_chunk(Wt, buf0, 0, Kpad, KC8, WROW, tid);

    for (int c = 0; c < nchunks; ++c) {
        _Float16* cur = (c & 1) ? buf1 : buf0;
        _Float16* nxt = (c & 1) ? buf0 : buf1;
        async_wait_all();
        __syncthreads();
        if (c + 1 < nchunks)
            stage_chunk(Wt, nxt, (c + 1) * Kc, Kpad, KC8, WROW, tid);

        const int kc  = c * Kc;
        const int nks = Kc >> 5;
        for (int ks = 0; ks < nks; ++ks) {
            v16h a = load_frag_a(
                A + (size_t)(row0 + lan16) * Kpad + (kc + ks * 32) + khi * 8);
            mma_row8(acc, a, cur + lan16 * WROW + ks * 32 + khi * 8, WROW);
        }
        __syncthreads();
    }

    #pragma unroll
    for (int nt = 0; nt < 8; ++nt) {
        int   n  = nt * 16 + lan16;
        float bv = bias[n];
        #pragma unroll
        for (int r = 0; r < 8; ++r) {
            int    m = row0 + khi * 8 + r;
            size_t o = (size_t)m * 128 + n;
            float  v = acc[nt][r] + bv;
            if (relu) v = fmaxf(v, 0.f);
            if (res)  v += res[o];
            if (outF) outF[o] = v;
            if (outH) outH[o] = (_Float16)v;
        }
    }
}

// ---------------------------------------------------------------------------
// Fused 2-layer MLP: out = relu(A@W1+b1)@W2+b2 [+res]
// T tile lives in LDS (each wave writes & reads its own 16 rows).
// Kpad must be a multiple of 128 (se:128, upd:640).
// ---------------------------------------------------------------------------
__global__ __launch_bounds__(256) void fused_mlp(
    const _Float16* __restrict__ A,
    const _Float16* __restrict__ W1t,   // 128 x Kpad f16 [n][k]
    const _Float16* __restrict__ W2t,   // 128 x 128  f16 [n][k]
    const float*    __restrict__ b1,
    const float*    __restrict__ b2,
    const float*    __restrict__ res,
    float*          __restrict__ outF,
    _Float16*       __restrict__ outH,
    int Kpad)
{
    extern __shared__ _Float16 smem[];

    const int tid   = threadIdx.x;
    const int lane  = tid & 31;
    const int wave  = tid >> 5;
    const int khi   = lane >> 4;
    const int lan16 = lane & 15;
    const int row0  = blockIdx.x * 128 + wave * 16;

    const int WROW    = 128 + WPAD;           // 136
    const int nchunks = Kpad / 128;
    _Float16* buf0 = smem;
    _Float16* buf1 = smem + 128 * WROW;
    _Float16* Tt   = smem + 2 * 128 * WROW;   // 128 x 136 f16 tile

    // ---- stage 1: acc = A @ W1 -------------------------------------------
    v8f acc[8] = {};
    stage_chunk(W1t, buf0, 0, Kpad, 16, WROW, tid);
    for (int c = 0; c < nchunks; ++c) {
        _Float16* cur = (c & 1) ? buf1 : buf0;
        _Float16* nxt = (c & 1) ? buf0 : buf1;
        async_wait_all();
        __syncthreads();
        if (c + 1 < nchunks)
            stage_chunk(W1t, nxt, (c + 1) * 128, Kpad, 16, WROW, tid);

        const int kc = c * 128;
        for (int ks = 0; ks < 4; ++ks) {
            v16h a = load_frag_a(
                A + (size_t)(row0 + lan16) * Kpad + (kc + ks * 32) + khi * 8);
            mma_row8(acc, a, cur + lan16 * WROW + ks * 32 + khi * 8, WROW);
        }
        __syncthreads();
    }

    // stage W2 (overlaps with T-tile write below)
    stage_chunk(W2t, buf0, 0, 128, 16, WROW, tid);

    // T = relu(acc + b1) -> LDS f16 (wave-private rows)
    const int mloc0 = wave * 16 + khi * 8;
    #pragma unroll
    for (int nt = 0; nt < 8; ++nt) {
        int   n  = nt * 16 + lan16;
        float bv = b1[n];
        #pragma unroll
        for (int r = 0; r < 8; ++r) {
            float t = fmaxf(acc[nt][r] + bv, 0.f);
            Tt[(mloc0 + r) * WROW + n] = (_Float16)t;
        }
    }
    async_wait_all();
    __syncthreads();

    // ---- stage 2: acc2 = T @ W2 ------------------------------------------
    v8f acc2[8] = {};
    for (int ks = 0; ks < 4; ++ks) {
        v16h a = load_frag_a(
            Tt + (wave * 16 + lan16) * WROW + ks * 32 + khi * 8);
        mma_row8(acc2, a, buf0 + lan16 * WROW + ks * 32 + khi * 8, WROW);
    }

    #pragma unroll
    for (int nt = 0; nt < 8; ++nt) {
        int   n  = nt * 16 + lan16;
        float bv = b2[n];
        #pragma unroll
        for (int r = 0; r < 8; ++r) {
            int    m = row0 + khi * 8 + r;
            size_t o = (size_t)m * 128 + n;
            float  v = acc2[nt][r] + bv;
            if (res)  v += res[o];
            if (outF) outF[o] = v;
            if (outH) outH[o] = (_Float16)v;
        }
    }
}

// ---------------------------------------------------------------------------
// Attention scores + softmax + messages + cat build (one block per batch item)
// ---------------------------------------------------------------------------
__global__ __launch_bounds__(256) void attn_msg(
    const float* __restrict__ h,     // ROWS x 128 f32
    const float* __restrict__ semb,  // BTOT x 128 f32
    const float* __restrict__ Wl,    // 384 x 4  (Wi|Wj|Ws)
    const float* __restrict__ bl,    // 4
    _Float16*    __restrict__ cat)   // ROWS x 640 f16
{
    __shared__ float hl[NAG][EMB];
    __shared__ float sl[EMB];
    __shared__ float ps[64][4];
    __shared__ float pss[HD][4];
    __shared__ float si[NAG][HD];
    __shared__ float sj[NAG][HD];
    __shared__ float ss[HD];
    __shared__ float at[NAG][NAG][HD];

    const int b   = blockIdx.x;
    const int tid = threadIdx.x;

    for (int idx = tid; idx < NAG * EMB; idx += 256) {
        int i = idx >> 7, e = idx & 127;
        float v = h[(size_t)(b * NAG + i) * EMB + e];
        hl[i][e] = v;
        cat[(size_t)(b * NAG + i) * KCAT + e] = (_Float16)v;
    }
    for (int idx = tid; idx < EMB; idx += 256)
        sl[idx] = semb[(size_t)b * EMB + idx];
    __syncthreads();

    {   // 64 dots (si:0..31, sj:32..63) x 4 partials of 32 elements
        int d = tid >> 2, q = tid & 3;
        int dd = d & 31, i = dd >> 2, hh = dd & 3;
        const float* base = (d < 32) ? Wl : (Wl + EMB * HD);
        float s = 0.f;
        for (int e = q * 32; e < q * 32 + 32; ++e)
            s += hl[i][e] * base[e * HD + hh];
        ps[d][q] = s;
    }
    if (tid >= 64 && tid < 64 + 16) {          // ss partials
        int t = tid - 64, hh = t >> 2, q = t & 3;
        const float* base = Wl + 2 * EMB * HD;
        float s = 0.f;
        for (int e = q * 32; e < q * 32 + 32; ++e)
            s += sl[e] * base[e * HD + hh];
        pss[hh][q] = s;
    }
    __syncthreads();
    if (tid < 64) {
        int dd = tid & 31, i = dd >> 2, hh = dd & 3;
        float s = ps[tid][0] + ps[tid][1] + ps[tid][2] + ps[tid][3];
        if (tid < 32) si[i][hh] = s; else sj[i][hh] = s;
    }
    if (tid >= 64 && tid < 64 + HD) {
        int hh = tid - 64;
        ss[hh] = pss[hh][0] + pss[hh][1] + pss[hh][2] + pss[hh][3];
    }
    __syncthreads();

    if (tid < 32) {                       // softmax over j
        int i = tid >> 2, hh = tid & 3;
        float base = si[i][hh] + ss[hh] + bl[hh];
        float sc[NAG], mx = -1e30f;
        for (int j = 0; j < NAG; ++j) { sc[j] = base + sj[j][hh]; mx = fmaxf(mx, sc[j]); }
        float sum = 0.f;
        for (int j = 0; j < NAG; ++j) { sc[j] = __expf(sc[j] - mx); sum += sc[j]; }
        float inv = 1.f / sum;
        for (int j = 0; j < NAG; ++j) at[i][j][hh] = sc[j] * inv;
    }
    __syncthreads();

    for (int idx = tid; idx < NAG * HD * EMB; idx += 256) {
        int i = idx >> 9, hh = (idx >> 7) & 3, e = idx & 127;
        float s = 0.f;
        #pragma unroll
        for (int j = 0; j < NAG; ++j) s += at[i][j][hh] * hl[j][e];
        cat[(size_t)(b * NAG + i) * KCAT + EMB + hh * EMB + e] = (_Float16)s;
    }
}

// ---------------------------------------------------------------------------
// final: q_tot[row] = dot(t[row,:], qp_W2) + qp_b2
// ---------------------------------------------------------------------------
__global__ __launch_bounds__(256) void qdot(const _Float16* __restrict__ t,
                                            const float* __restrict__ W2,
                                            const float* __restrict__ b2,
                                            float* __restrict__ out)
{
    int r = blockIdx.x * 256 + threadIdx.x;
    if (r >= BTOT) return;
    float s = b2[0];
    for (int e = 0; e < EMB; ++e) s += (float)t[(size_t)r * EMB + e] * W2[e];
    out[r] = s;
}

// ---------------------------------------------------------------------------
// host launch
// ---------------------------------------------------------------------------
extern "C" void kernel_launch(void* const* d_in, const int* in_sizes, int n_in,
                              void* d_out, int out_size, void* d_ws, size_t ws_size,
                              hipStream_t stream)
{
    (void)in_sizes; (void)n_in; (void)out_size; (void)ws_size;

    const float* agent_qs  = (const float*)d_in[0];
    const float* states    = (const float*)d_in[1];
    const float* agent_obs = (const float*)d_in[2];
    const float* node_W    = (const float*)d_in[3];
    const float* node_b    = (const float*)d_in[4];
    const float* se_W1     = (const float*)d_in[5];
    const float* se_b1     = (const float*)d_in[6];
    const float* se_W2     = (const float*)d_in[7];
    const float* se_b2     = (const float*)d_in[8];
    const float* attn_W    = (const float*)d_in[9];
    const float* attn_b    = (const float*)d_in[10];
    const float* upd_W1    = (const float*)d_in[11];
    const float* upd_b1    = (const float*)d_in[12];
    const float* upd_W2    = (const float*)d_in[13];
    const float* upd_b2    = (const float*)d_in[14];
    const float* qp_W1     = (const float*)d_in[15];
    const float* qp_b1     = (const float*)d_in[16];
    const float* qp_W2     = (const float*)d_in[17];
    const float* qp_b2     = (const float*)d_in[18];
    float* out             = (float*)d_out;

    char* ws = (char*)d_ws;
    size_t off = 0;
    auto take = [&](size_t bytes) { char* p = ws + off; off = (off + bytes + 255) & ~(size_t)255; return p; };
    _Float16* nf    = (_Float16*)take((size_t)ROWS * KNF * 2);
    _Float16* sth   = (_Float16*)take((size_t)BTOT * SD  * 2);
    float*    semb  = (float*)   take((size_t)BTOT * EMB * 4);
    float*    h32   = (float*)   take((size_t)ROWS * EMB * 4);
    _Float16* h16   = (_Float16*)take((size_t)ROWS * EMB * 2);
    _Float16* cat   = (_Float16*)take((size_t)ROWS * KCAT * 2);
    _Float16* t_qp  = (_Float16*)take((size_t)BTOT * EMB * 2);
    // pre-transposed f16 weights
    _Float16* w_node = (_Float16*)take((size_t)128 * KNF  * 2);
    _Float16* w_se1  = (_Float16*)take((size_t)128 * SD   * 2);
    _Float16* w_se2  = (_Float16*)take((size_t)128 * EMB  * 2);
    _Float16* w_u1   = (_Float16*)take((size_t)NL * 128 * KCAT * 2);
    _Float16* w_u2   = (_Float16*)take((size_t)NL * 128 * EMB  * 2);
    _Float16* w_qp   = (_Float16*)take((size_t)128 * KQP  * 2);

    const size_t shm96   = 2u * 128 * (KNF + WPAD) * sizeof(_Float16);  //  53248
    const size_t shm128  = 2u * 128 * (128 + WPAD) * sizeof(_Float16);  //  69632
    const size_t shmfuse = 3u * 128 * (128 + WPAD) * sizeof(_Float16);  // 104448

    // 0. one-time weight transforms
    wprep<<<(128*KNF  + 255)/256, 256, 0, stream>>>(node_W, w_node, 65,  KNF);
    wprep<<<(128*SD   + 255)/256, 256, 0, stream>>>(se_W1,  w_se1, SD,  SD);
    wprep<<<(128*EMB  + 255)/256, 256, 0, stream>>>(se_W2,  w_se2, EMB, EMB);
    for (int l = 0; l < NL; ++l) {
        wprep<<<(128*KCAT + 255)/256, 256, 0, stream>>>(
            upd_W1 + (size_t)l*KCAT*EMB, w_u1 + (size_t)l*128*KCAT, KCAT, KCAT);
        wprep<<<(128*EMB  + 255)/256, 256, 0, stream>>>(
            upd_W2 + (size_t)l*EMB*EMB,  w_u2 + (size_t)l*128*EMB,  EMB,  EMB);
    }
    wprep<<<(128*KQP + 255)/256, 256, 0, stream>>>(qp_W1, w_qp, KQP, KQP);

    // 1. featurize
    prep_nf<<<(ROWS * KNF + 255) / 256, 256, 0, stream>>>(agent_qs, agent_obs, nf);
    prep_st<<<(BTOT * SD + 255) / 256, 256, 0, stream>>>(states, sth);

    // 2. h0 = nf @ node_W + b
    gemm128<<<ROWS / 128, 256, shm96, stream>>>(nf, w_node, node_b,
                                                nullptr, h32, nullptr, KNF, 0);
    // 3. state_emb = relu(st@W1+b1)@W2+b2   (fused)
    fused_mlp<<<BTOT / 128, 256, shmfuse, stream>>>(sth, w_se1, w_se2,
                                                    se_b1, se_b2,
                                                    nullptr, semb, nullptr, SD);

    // 4. GNN layers: h += relu(cat@W1+b1)@W2+b2  (fused)
    for (int l = 0; l < NL; ++l) {
        attn_msg<<<BTOT, 256, 0, stream>>>(h32, semb,
                                           attn_W + (size_t)l * 3 * EMB * HD,
                                           attn_b + (size_t)l * HD, cat);
        fused_mlp<<<ROWS / 128, 256, shmfuse, stream>>>(cat,
                                           w_u1 + (size_t)l * 128 * KCAT,
                                           w_u2 + (size_t)l * 128 * EMB,
                                           upd_b1 + (size_t)l * EMB,
                                           upd_b2 + (size_t)l * EMB,
                                           h32, h32, h16, KCAT);
    }

    // 5. q_tot = relu(h_flat @ qp_W1 + b1) @ qp_W2 + b2
    gemm128<<<BTOT / 128, 256, shm128, stream>>>(h16, w_qp, qp_b1,
                                                 nullptr, nullptr, t_qp, KQP, 1);
    qdot<<<(BTOT + 255) / 256, 256, 0, stream>>>(t_qp, qp_W2, qp_b2, out);
}
